// WDiscriminator_2353642078846
// MI455X (gfx1250) — matrix operand order repaired and checked by reading, not compiled, not run on hardware
//
#include <hip/hip_runtime.h>
#include <hip/hip_bf16.h>

typedef __attribute__((ext_vector_type(16))) __bf16 v16bf;
typedef __attribute__((ext_vector_type(8)))  __bf16 v8bf;
typedef __attribute__((ext_vector_type(4)))  __bf16 v4bf;
typedef __attribute__((ext_vector_type(8)))  float  v8f;
typedef __attribute__((ext_vector_type(4)))  int    v4i;

#define NEG_SLOPE 0.2f
#define DIN 256
#define DH  512

__device__ __forceinline__ float leaky(float x) { return x > 0.f ? x : NEG_SLOPE * x; }

// ---- CDNA5 async global->LDS staging (ASYNCcnt path), feature-guarded ----
#if __has_builtin(__builtin_amdgcn_global_load_async_to_lds_b128) && \
    __has_builtin(__builtin_amdgcn_s_wait_asynccnt)
#define HAVE_ASYNC_LDS 1
#else
#define HAVE_ASYNC_LDS 0
#endif

// addrspace-qualified pointee typedefs for the async builtins
typedef __attribute__((address_space(1))) v4i gbl_v4i;  // prints as "__device__"
typedef __attribute__((address_space(3))) v4i lds_v4i;  // prints as "__shared__"

__device__ __forceinline__ gbl_v4i* as_gbl_v4i(const void* p) {
  return (gbl_v4i*)(uintptr_t)p;  // also strips const (builtin takes non-const)
}
__device__ __forceinline__ lds_v4i* as_lds_v4i(void* p) {
  // LDS generic addresses carry the LDS byte offset in addr[31:0] (ISA 10.2).
  return (lds_v4i*)(unsigned)(uintptr_t)p;
}

// ---------------------------------------------------------------- utilities

__global__ void fill_f32(float* p, float v, long n) {
  long i = (long)blockIdx.x * blockDim.x + threadIdx.x;
  long s = (long)gridDim.x * blockDim.x;
  for (; i < n; i += s) p[i] = v;
}

__global__ void cvt_f32_bf16(const float* in, __bf16* out, long n) {
  long i = (long)blockIdx.x * blockDim.x + threadIdx.x;
  long s = (long)gridDim.x * blockDim.x;
  for (; i < n; i += s) out[i] = (__bf16)in[i];
}

__global__ void deg_count(const long long* ei, float* deg, int E) {
  int e = blockIdx.x * blockDim.x + threadIdx.x;
  if (e < E) atomicAdd(&deg[(int)ei[(long)E + e]], 1.0f);  // dst row
}

__global__ void dinv_kernel(const float* deg, float* dinv, int N) {
  int i = blockIdx.x * blockDim.x + threadIdx.x;
  if (i < N) {
    float d = deg[i];
    dinv[i] = d > 0.f ? rsqrtf(d) : 0.f;
  }
}

// ------------------------------------------------- WMMA bf16 GEMM (generic)
// C[M,N](bf16) = epilogue( A[M,K] @ B[K,N] )
// AMODE 0: A is bf16, staged raw (async global->LDS when available).
// AMODE 1: A is f32; prologue a = leaky(a + abias[k]) fused into LDS stage.
// Epilogue: +cbias[n] (if non-null), leaky if relu_out, store bf16.

#define BM 64
#define BN 128
#define BK 32

template <int AMODE>
__global__ __launch_bounds__(256) void gemm_wmma(
    const void* __restrict__ Aptr, const __bf16* __restrict__ B,
    __bf16* __restrict__ C, const float* __restrict__ abias,
    const float* __restrict__ cbias, int relu_out, int M, int N, int K) {
  __shared__ alignas(32) __bf16 sA[BM * BK];      // row-major [m][k]
  __shared__ alignas(32) __bf16 sB[BK * BN];      // row-major [k][n]

  const int tid  = threadIdx.x;
  const int wid  = tid >> 5;
  const int lane = tid & 31;
  const int m0   = blockIdx.x * BM;
  const int n0   = blockIdx.y * BN;

  // 8 waves as 2 (M) x 4 (N); each wave owns a 32x32 tile = 2x2 WMMA tiles
  const int wM = (wid >> 2) * 32;
  const int wN = (wid & 3) * 32;

  v8f acc[2][2];
  for (int mi = 0; mi < 2; mi++)
    for (int ni = 0; ni < 2; ni++)
      for (int j = 0; j < 8; j++) acc[mi][ni][j] = 0.f;

  const int lrow = lane & 15;            // A row / B col / D col within tile
  const int kbA  = (lane >> 4) * 8;      // A frag: K run base (0 or 8)
  const int kbB  = (lane >> 4) * 16;     // unused (B frag also uses kbA)

  for (int k0 = 0; k0 < K; k0 += BK) {
    // ---- stage A tile (64x32): thread -> row tid>>2, 8 elems at (tid&3)*8
    {
      const int r  = tid >> 2;
      const int c  = (tid & 3) * 8;
      const int gm = m0 + r;
      if (AMODE == 0) {
        const __bf16* A = (const __bf16*)Aptr;
        if (gm < M) {
#if HAVE_ASYNC_LDS
          __builtin_amdgcn_global_load_async_to_lds_b128(
              as_gbl_v4i(&A[(long)gm * K + k0 + c]),
              as_lds_v4i(&sA[r * BK + c]), 0, 0);
#else
          *reinterpret_cast<v8bf*>(&sA[r * BK + c]) =
              *reinterpret_cast<const v8bf*>(&A[(long)gm * K + k0 + c]);
#endif
        } else {
          v8bf z;
          for (int j = 0; j < 8; j++) z[j] = (__bf16)0.f;
          *reinterpret_cast<v8bf*>(&sA[r * BK + c]) = z;
        }
      } else {
        const float* A = (const float*)Aptr;
        v8bf v;
        if (gm < M) {
          for (int j = 0; j < 8; j++) {
            float x = A[(long)gm * K + k0 + c + j] + abias[k0 + c + j];
            v[j] = (__bf16)leaky(x);
          }
        } else {
          for (int j = 0; j < 8; j++) v[j] = (__bf16)0.f;
        }
        *reinterpret_cast<v8bf*>(&sA[r * BK + c]) = v;
      }
    }
    // ---- stage B tile (32x128) row-major: thread -> k tid>>3, 16 cols
    {
      const int kk = tid >> 3;
      const int c  = (tid & 7) * 16;
      const __bf16* gsrc = &B[(long)(k0 + kk) * N + n0 + c];
      __bf16* ldst = &sB[kk * BN + c];
#if HAVE_ASYNC_LDS
      __builtin_amdgcn_global_load_async_to_lds_b128(as_gbl_v4i(gsrc),
                                                     as_lds_v4i(ldst), 0, 0);
      __builtin_amdgcn_global_load_async_to_lds_b128(as_gbl_v4i(gsrc + 8),
                                                     as_lds_v4i(ldst + 8), 0, 0);
#else
      *reinterpret_cast<v8bf*>(ldst) = *reinterpret_cast<const v8bf*>(gsrc);
      *reinterpret_cast<v8bf*>(ldst + 8) =
          *reinterpret_cast<const v8bf*>(gsrc + 8);
#endif
    }
#if HAVE_ASYNC_LDS
    __builtin_amdgcn_s_wait_asynccnt(0);  // our LDS writes landed
#endif
    __syncthreads();

    for (int mi = 0; mi < 2; mi++) {
      // A fragment per ISA 7.12.2: lane<16 -> K 0-7 & 16-23; lane>=16 -> 8-15 & 24-31
      const v8bf a0 = *reinterpret_cast<const v8bf*>(
          &sA[(wM + mi * 16 + lrow) * BK + kbA]);
      const v8bf a1 = *reinterpret_cast<const v8bf*>(
          &sA[(wM + mi * 16 + lrow) * BK + 16 + kbA]);
      v16bf a;
      for (int j = 0; j < 8; j++) { a[j] = a0[j]; a[8 + j] = a1[j]; }
      for (int ni = 0; ni < 2; ni++) {
        // B fragment: col = lane&15; 16 K values gathered from row-major sB
        const int ncol = wN + ni * 16 + lrow;
        v16bf b;
#pragma unroll
        for (int j = 0; j < 16; j++) b[j] = sB[((j < 8 ? 0 : 8) + kbA + j) * BN + ncol];
        acc[mi][ni] = __builtin_amdgcn_wmma_f32_16x16x32_bf16(
            false, a, false, b, (short)0, acc[mi][ni], false, false);
      }
    }
    __syncthreads();
  }

  // ---- epilogue: D layout = VGPR r -> M = r (lane<16) / 8+r (lane>=16), N = lane&15
  for (int mi = 0; mi < 2; mi++) {
    for (int ni = 0; ni < 2; ni++) {
      const int ncol = n0 + wN + ni * 16 + lrow;
      const float bb = cbias ? cbias[ncol] : 0.f;
      for (int r = 0; r < 8; r++) {
        const int gm = m0 + wM + mi * 16 + ((lane < 16) ? r : 8 + r);
        if (gm < M) {
          float v = acc[mi][ni][r] + bb;
          if (relu_out) v = leaky(v);
          C[(long)gm * N + ncol] = (__bf16)v;
        }
      }
    }
  }
}

// -------------------------------------------- edge scatter (L2 f32 atomics)
// agg[dst,:] += h0[src,:] * dinv[src]*dinv[dst]; edges E..E+N-1 are self-loops.
__global__ __launch_bounds__(128) void scatter_kernel(
    const __bf16* __restrict__ h0, const long long* __restrict__ ei,
    const float* __restrict__ dinv, float* __restrict__ agg, int E, int N) {
  const int f = threadIdx.x * 4;  // 128 threads * 4 feats = 512
  const long total = (long)E + N;
  for (long e = blockIdx.x; e < total; e += gridDim.x) {
    __builtin_prefetch(&ei[e + gridDim.x], 0, 0);  // global_prefetch_b8
    int s, d;
    if (e < E) { s = (int)ei[e]; d = (int)ei[(long)E + e]; }
    else       { s = d = (int)(e - E); }
    const float norm = dinv[s] * dinv[d];
    const v4bf hv = *reinterpret_cast<const v4bf*>(&h0[(long)s * DH + f]);
    float* ap = &agg[(long)d * DH + f];
    atomicAdd(&ap[0], (float)hv[0] * norm);
    atomicAdd(&ap[1], (float)hv[1] * norm);
    atomicAdd(&ap[2], (float)hv[2] * norm);
    atomicAdd(&ap[3], (float)hv[3] * norm);
  }
}

// --------------------------------------------------- final matvec + bias
__global__ __launch_bounds__(256) void rowdot_kernel(
    const __bf16* __restrict__ h2, const float* __restrict__ w,
    const float* __restrict__ b, float* __restrict__ y, int Nrows) {
  const int row  = blockIdx.x * 8 + (threadIdx.x >> 5);
  const int lane = threadIdx.x & 31;
  if (row >= Nrows) return;
  float s = 0.f;
  for (int k = lane; k < DH; k += 32)
    s += (float)h2[(long)row * DH + k] * w[k];
  for (int off = 16; off > 0; off >>= 1) s += __shfl_xor(s, off, 32);
  if (lane == 0) y[row] = s + b[0];
}

// ---------------------------------------------------------------- launcher

extern "C" void kernel_launch(void* const* d_in, const int* in_sizes, int n_in,
                              void* d_out, int out_size, void* d_ws,
                              size_t ws_size, hipStream_t stream) {
  (void)n_in; (void)out_size; (void)ws_size;
  const float*     x  = (const float*)d_in[0];
  const long long* ei = (const long long*)d_in[1];  // int64 [2,E]
  const float*     Wg = (const float*)d_in[2];
  const float*     bg = (const float*)d_in[3];
  const float*     W2 = (const float*)d_in[4];
  const float*     b2 = (const float*)d_in[5];
  const float*     Wo = (const float*)d_in[6];
  const float*     bo = (const float*)d_in[7];
  float*           out = (float*)d_out;

  const int N = in_sizes[0] / DIN;
  const int E = in_sizes[1] / 2;

  char*  ws  = (char*)d_ws;
  size_t off = 0;
  auto carve = [&](size_t bytes) -> void* {
    void* p = ws + off;
    off = (off + bytes + 255) & ~(size_t)255;
    return p;
  };
  float*  deg  = (float*)carve((size_t)N * 4);
  float*  dinv = (float*)carve((size_t)N * 4);
  __bf16* xb   = (__bf16*)carve((size_t)N * DIN * 2);
  __bf16* wgb  = (__bf16*)carve((size_t)DIN * DH * 2);
  __bf16* w2b  = (__bf16*)carve((size_t)DH * DH * 2);
  __bf16* h0   = (__bf16*)carve((size_t)N * DH * 2);
  float*  agg  = (float*)carve((size_t)N * DH * 4);
  __bf16* h2   = (__bf16*)carve((size_t)N * DH * 2);

  // 1) degrees (self-loop seeds deg=1), zero accumulator
  fill_f32<<<512, 256, 0, stream>>>(deg, 1.0f, (long)N);
  fill_f32<<<4096, 256, 0, stream>>>(agg, 0.0f, (long)N * DH);
  deg_count<<<(E + 255) / 256, 256, 0, stream>>>(ei, deg, E);
  dinv_kernel<<<(N + 255) / 256, 256, 0, stream>>>(deg, dinv, N);

  // 2) bf16 staging of GEMM operands
  cvt_f32_bf16<<<4096, 256, 0, stream>>>(x, xb, (long)N * DIN);
  cvt_f32_bf16<<<256, 256, 0, stream>>>(Wg, wgb, (long)DIN * DH);
  cvt_f32_bf16<<<512, 256, 0, stream>>>(W2, w2b, (long)DH * DH);

  // 3) h0 = x @ W_gcn   (bias deferred until after aggregation, per GCNConv)
  {
    dim3 grid((N + BM - 1) / BM, DH / BN);
    gemm_wmma<0><<<grid, 256, 0, stream>>>(xb, wgb, h0, nullptr, nullptr, 0,
                                           N, DH, DIN);
  }

  // 4) agg[dst] += h0[src] * dinv[src]*dinv[dst]  (+ self loops)
  scatter_kernel<<<8192, 128, 0, stream>>>(h0, ei, dinv, agg, E, N);

  // 5) h2 = leaky( leaky(agg + b_gcn) @ W2 + b2 )   (prologue+epilogue fused)
  {
    dim3 grid((N + BM - 1) / BM, DH / BN);
    gemm_wmma<1><<<grid, 256, 0, stream>>>(agg, w2b, h2, bg, b2, 1,
                                           N, DH, DH);
  }

  // 6) out = h2 @ W_out + b_out
  rowdot_kernel<<<(N + 7) / 8, 256, 0, stream>>>(h2, Wo, bo, out, N);
}
